// _ChannelAttentionModule_20426864460440
// MI455X (gfx1250) — matrix operand order repaired and checked by reading, not compiled, not run on hardware
//
#include <hip/hip_runtime.h>
#include <hip/hip_bf16.h>

// ---------------------------------------------------------------------------
// DANet channel attention for MI455X (gfx1250, wave32, WMMA).
//   feat  = bf16(x)            [B][C][N]
//   featT = bf16(x)^T          [B][N][C]   (for contiguous B-fragments in GEMM2)
//   S     = feat . feat^T      (v_wmma_f32_16x16x32_bf16, K=N=1024)
//   P     = softmax(rowmax(S)-S) == exp(rowmin(S)-S)/sum  (fused in LDS)
//   out   = beta * (P . feat) + x          (v_wmma + fused epilogue)
// ---------------------------------------------------------------------------

typedef __attribute__((ext_vector_type(16))) __bf16 v16bf;
typedef __attribute__((ext_vector_type(8)))  __bf16 v8bf;
typedef __attribute__((ext_vector_type(8)))  float  v8f;

#define B_ 32
#define C_ 512
#define N_ 1024

union AFrag { v16bf v; v8bf h[2]; };

// ---------------------------------------------------------------------------
// Kernel 1: fp32 -> bf16 convert + transpose (32x32 tiles via LDS)
// ---------------------------------------------------------------------------
__global__ __launch_bounds__(256) void k_convert_transpose(
    const float* __restrict__ x, __bf16* __restrict__ feat,
    __bf16* __restrict__ featT) {
  __shared__ float tile[32][33];  // +1 pad: no LDS bank conflicts on transpose
  const int blk = blockIdx.x;
  const int nt = blk & 31;         // N_/32 tiles
  const int ct = (blk >> 5) & 15;  // C_/32 tiles
  const int b  = blk >> 9;
  const int tid = threadIdx.x;
  const int tx = tid & 31;
  const int ty = tid >> 5;         // 0..7
  const int c0 = ct * 32, n0 = nt * 32;
  const size_t xbase = (size_t)b * C_ * N_;
#pragma unroll
  for (int r = 0; r < 4; ++r) {
    const int cl = ty + 8 * r;
    const size_t idx = xbase + (size_t)(c0 + cl) * N_ + n0 + tx;
    const float v = x[idx];
    tile[cl][tx] = v;
    feat[idx] = (__bf16)v;  // coalesced along N
  }
  __syncthreads();
  const size_t tbase = (size_t)b * N_ * C_;
#pragma unroll
  for (int r = 0; r < 4; ++r) {
    const int nl = ty + 8 * r;
    featT[tbase + (size_t)(n0 + nl) * C_ + c0 + tx] =
        (__bf16)tile[tx][nl];  // coalesced along C
  }
}

// ---------------------------------------------------------------------------
// Kernel 2: fused Gram (WMMA) + reversed softmax.
// One WG (8 waves) owns a full 16x512 row-block of S for one batch.
// Wave w computes cols [64w, 64w+64) with 4 f32 accumulators; K loops over N.
// A-frag (16x32 bf16): lane l<16 -> row m0+l, K 0..15; lanes 16..31
// hold K 16..31 (two 16B loads).  B-frag = F^T tile: lane col n is a
// fixed F row, K contiguous -> one 32B load.  Same contiguous K order as A.
// ---------------------------------------------------------------------------
__global__ __launch_bounds__(256) void k_gram_softmax(
    const __bf16* __restrict__ feat, __bf16* __restrict__ attnP) {
  __shared__ float srow[16][512];  // 32 KB score row-block
  __shared__ float red[16][16];
  __shared__ float stat[16];

  const int tid  = threadIdx.x;
  const int lane = tid & 31;
  const int wave = tid >> 5;   // 0..7
  const int half = lane >> 4;  // lane half selects K sub-range
  const int lcol = lane & 15;

  const int blk = blockIdx.x;
  const int mt  = blk & 31;    // C_/16 row tiles
  const int b   = blk >> 5;
  const int m0  = mt * 16;

  const __bf16* F = feat + (size_t)b * C_ * N_;
  const __bf16* arow = F + (size_t)(m0 + lcol) * N_;
  const __bf16* brow0 = F + (size_t)(wave * 64 + lcol) * N_;
  const int d0 = wave * 64;

  v8f acc[4] = {};
  for (int k = 0; k < N_; k += 32) {
    // WGP-scope prefetch (locality 3) of the next K-chunk into near cache.
    __builtin_prefetch(arow + k + 64, 0, 3);
    __builtin_prefetch(brow0 + k + 64, 0, 3);
    AFrag afr;
    afr.h[0] = *(const v8bf*)(arow + k + 16 * half);
    afr.h[1] = *(const v8bf*)(arow + k + 16 * half + 8);
#pragma unroll
    for (int t = 0; t < 4; ++t) {
      const v16bf bfr =
          *(const v16bf*)(F + (size_t)(d0 + t * 16 + lcol) * N_ + k + 16 * half);
      acc[t] = __builtin_amdgcn_wmma_f32_16x16x32_bf16(
          false, afr.v, false, bfr, (short)0, acc[t], false, false);
    }
  }

  // Spill accumulators: elem i, lane l -> row i+8*half, col lane&15 (ISA C/D).
#pragma unroll
  for (int t = 0; t < 4; ++t) {
    const int col = d0 + t * 16 + lcol;
#pragma unroll
    for (int i = 0; i < 8; ++i) srow[i + 8 * half][col] = acc[t][i];
  }
  __syncthreads();

  // softmax(rowmax - S) == exp(rowmin - S) / sum(exp(rowmin - S))
  const int row = tid >> 4;  // 16 threads per row
  const int seg = tid & 15;
  float mn = 3.0e38f;
#pragma unroll
  for (int j = 0; j < 32; ++j) mn = fminf(mn, srow[row][seg + 16 * j]);
  red[row][seg] = mn;
  __syncthreads();
  if (tid < 16) {
    float m = red[tid][0];
#pragma unroll
    for (int j = 1; j < 16; ++j) m = fminf(m, red[tid][j]);
    stat[tid] = m;
  }
  __syncthreads();
  const float rmin = stat[row];
  float ps = 0.0f;
#pragma unroll
  for (int j = 0; j < 32; ++j) {
    const int col = seg + 16 * j;
    const float e = __expf(rmin - srow[row][col]);
    srow[row][col] = e;
    ps += e;
  }
  __syncthreads();
  red[row][seg] = ps;
  __syncthreads();
  if (tid < 16) {
    float s = 0.0f;
#pragma unroll
    for (int j = 0; j < 16; ++j) s += red[tid][j];
    stat[tid] = 1.0f / s;
  }
  __syncthreads();
  const float inv = stat[row];
  __bf16* P = attnP + (size_t)b * C_ * C_ + (size_t)(m0 + row) * C_;
#pragma unroll
  for (int j = 0; j < 32; ++j) {
    const int col = seg + 16 * j;
    P[col] = (__bf16)(srow[row][col] * inv);  // coalesced bf16 store
  }
}

// ---------------------------------------------------------------------------
// Kernel 3: E = P . F via WMMA (B-fragments from featT => contiguous loads),
// fused epilogue out = beta*E + x.  WG = 16 rows x 512 cols, wave = 16x64.
// ---------------------------------------------------------------------------
__global__ __launch_bounds__(256) void k_attn_apply(
    const __bf16* __restrict__ attnP, const __bf16* __restrict__ featT,
    const float* __restrict__ x, const float* __restrict__ beta_p,
    float* __restrict__ out) {
  const int tid  = threadIdx.x;
  const int lane = tid & 31;
  const int wave = tid >> 5;
  const int half = lane >> 4;
  const int lcol = lane & 15;

  const int blk = blockIdx.x;
  const int nh = blk & 1;          // which 512-col half
  const int mt = (blk >> 1) & 31;  // C_/16 row tiles
  const int b  = blk >> 6;

  const int m0 = mt * 16;
  const int n0 = nh * 512 + wave * 64;

  const __bf16* P  = attnP + (size_t)b * C_ * C_;
  const __bf16* FT = featT + (size_t)b * N_ * C_;
  const __bf16* arow  = P + (size_t)(m0 + lcol) * C_;
  const __bf16* brow0 = FT + (size_t)(n0 + lcol) * C_;

  v8f acc[4] = {};
  for (int k = 0; k < C_; k += 32) {
    __builtin_prefetch(arow + k + 64, 0, 3);   // WGP-scope near-cache prefetch
    __builtin_prefetch(brow0 + k + 64, 0, 3);
    AFrag afr;
    afr.h[0] = *(const v8bf*)(arow + k + 16 * half);
    afr.h[1] = *(const v8bf*)(arow + k + 16 * half + 8);
#pragma unroll
    for (int t = 0; t < 4; ++t) {
      const v16bf bfr =
          *(const v16bf*)(FT + (size_t)(n0 + t * 16 + lcol) * C_ + k + 16 * half);
      acc[t] = __builtin_amdgcn_wmma_f32_16x16x32_bf16(
          false, afr.v, false, bfr, (short)0, acc[t], false, false);
    }
  }

  const float beta = beta_p[0];
  const size_t xb = (size_t)b * C_ * N_;
#pragma unroll
  for (int t = 0; t < 4; ++t) {
#pragma unroll
    for (int i = 0; i < 8; ++i) {
      const int row = m0 + i + 8 * half;
      const int col = n0 + t * 16 + lcol;
      const size_t idx = xb + (size_t)row * N_ + col;
      out[idx] = beta * acc[t][i] + x[idx];
    }
  }
}

// ---------------------------------------------------------------------------
extern "C" void kernel_launch(void* const* d_in, const int* in_sizes, int n_in,
                              void* d_out, int out_size, void* d_ws,
                              size_t ws_size, hipStream_t stream) {
  (void)in_sizes; (void)n_in; (void)out_size; (void)ws_size;
  const float* x    = (const float*)d_in[0];
  const float* beta = (const float*)d_in[1];
  float* out = (float*)d_out;

  char* ws = (char*)d_ws;
  __bf16* feat  = (__bf16*)(ws);                          // 32 MB
  __bf16* featT = (__bf16*)(ws + (size_t)33554432);       // 32 MB
  __bf16* attnP = (__bf16*)(ws + (size_t)67108864);       // 16 MB

  k_convert_transpose<<<B_ * (C_ / 32) * (N_ / 32), 256, 0, stream>>>(x, feat,
                                                                     featT);
  k_gram_softmax<<<B_ * (C_ / 16), 256, 0, stream>>>(feat, attnP);
  k_attn_apply<<<B_ * (C_ / 16) * (N_ / 512), 256, 0, stream>>>(attnP, featT, x,
                                                                beta, out);
}
